// POS_45595372814587
// MI455X (gfx1250) — matrix operand order repaired and checked by reading, not compiled, not run on hardware
//
#include <hip/hip_runtime.h>

// ---------------------------------------------------------------------------
// Types
// ---------------------------------------------------------------------------
typedef __bf16 bf16;
typedef __bf16 bf16x8  __attribute__((ext_vector_type(8)));
typedef __bf16 v16bf   __attribute__((ext_vector_type(16)));
typedef float  v8f     __attribute__((ext_vector_type(8)));
typedef unsigned int u32x4 __attribute__((ext_vector_type(4)));

// pointer types for the gfx1250 async global->LDS builtins
// (signature per hipcc diagnostic: pointers to int4 in AS(1)/AS(3))
typedef int v4i __attribute__((__vector_size__(4 * sizeof(int))));
typedef __attribute__((address_space(1))) v4i* as1_v4i;
typedef __attribute__((address_space(3))) v4i* as3_v4i;

// gfx1250 async global->LDS path (ASYNCcnt), with sync fallback
#if defined(__HIP_DEVICE_COMPILE__) && __has_builtin(__builtin_amdgcn_global_load_async_to_lds_b128)
#define USE_ASYNC_LDS 1
#else
#define USE_ASYNC_LDS 0
#endif

__device__ __forceinline__ void wait_async0() {
#if defined(__HIP_DEVICE_COMPILE__)
#if __has_builtin(__builtin_amdgcn_s_wait_asynccnt)
  __builtin_amdgcn_s_wait_asynccnt(0);
#else
  asm volatile("s_wait_asynccnt 0x0" ::: "memory");
#endif
#endif
}

__device__ __forceinline__ void sched_fence() {
#if defined(__HIP_DEVICE_COMPILE__)
#if __has_builtin(__builtin_amdgcn_sched_barrier)
  __builtin_amdgcn_sched_barrier(0);   // keep fragment loads ahead of WMMAs
#endif
#endif
}

// ---------------------------------------------------------------------------
// fp32 -> bf16 conversion (vectorized, n divisible by 4)
// ---------------------------------------------------------------------------
__global__ void cvt_f32_to_bf16(const float* __restrict__ in,
                                bf16* __restrict__ out, int n4) {
  int i = blockIdx.x * blockDim.x + threadIdx.x;
  if (i >= n4) return;
  float4 v = ((const float4*)in)[i];
  bf16* o = out + (size_t)i * 4;
  o[0] = (bf16)v.x; o[1] = (bf16)v.y; o[2] = (bf16)v.z; o[3] = (bf16)v.w;
}

// ---------------------------------------------------------------------------
// Tiny fp32 GEMM: out[m,n] = dot(A[m,:], W[n,:]) + bias[n]
// rows m in [Mreal, Mtot) use A == 0 (-> out = bias), used for tproj row 45
// ---------------------------------------------------------------------------
__global__ void small_mm(const float* __restrict__ A, const float* __restrict__ W,
                         const float* __restrict__ bias, float* __restrict__ out,
                         int Mreal, int Mtot, int N, int K) {
  int idx = blockIdx.x * blockDim.x + threadIdx.x;
  if (idx >= Mtot * N) return;
  int m = idx / N, n = idx % N;
  float s = 0.f;
  if (m < Mreal) {
    const float4* a = (const float4*)(A + (size_t)m * K);
    const float4* w = (const float4*)(W + (size_t)n * K);
    for (int k = 0; k < K / 4; ++k) {
      float4 av = a[k], wv = w[k];
      s += av.x * wv.x + av.y * wv.y + av.z * wv.z + av.w * wv.w;
    }
  }
  out[idx] = s + bias[n];
}

// ---------------------------------------------------------------------------
// Tiled bf16 WMMA GEMM:  C[M,N] = A[M,K] @ W[N,K]^T  (+bias, +resid, relu)
// block tile 128x128, 8 waves of 64x32, K-tile 64 (2 x k32 WMMA steps)
// Double-buffered LDS; async global->LDS pipeline:
//   wait_asynccnt(0); barrier; issue async loads for tile k+1; WMMA on tile k
// ---------------------------------------------------------------------------
constexpr int BM = 128, BN = 128, KT = 64;
constexpr int AST = 72;  // LDS row stride in bf16 (144B: 16B aligned, bank-spread)

template <typename OutT, bool BIAS, bool RELU, bool RESID>
__global__ __launch_bounds__(256)
void gemm_bf16_wmma(const bf16* __restrict__ A, const bf16* __restrict__ Bw,
                    const float* __restrict__ bias, const float* __restrict__ resid,
                    OutT* __restrict__ C, int M, int N, int K) {
  __shared__ bf16 As[2][BM * AST];
  __shared__ bf16 Bs[2][BN * AST];

  const int tid  = threadIdx.x;
  const int lane = tid & 31;
  const int wave = tid >> 5;
  const int wm = wave >> 2;      // 0..1  (M direction, 64 rows each)
  const int wn = wave & 3;       // 0..3  (N direction, 32 cols each)
  const int hl  = lane >> 4;     // half-wave select
  const int l16 = lane & 15;
  const int bm = blockIdx.y * BM;
  const int bn = blockIdx.x * BN;
  const size_t Kz = (size_t)K;

  // staging chunk coordinates: 1024 x 16B chunks per tile, 4 per thread
  int rr[4], cc[4];
#pragma unroll
  for (int i = 0; i < 4; ++i) {
    int c = tid + i * 256;
    rr[i] = c >> 3;
    cc[i] = (c & 7) * 8;
  }

  auto stage = [&](int p, int kt) {
#pragma unroll
    for (int i = 0; i < 4; ++i) {
      const bf16* ga = &A [(size_t)(bm + rr[i]) * Kz + kt + cc[i]];
      const bf16* gb = &Bw[(size_t)(bn + rr[i]) * Kz + kt + cc[i]];
      bf16* la = &As[p][rr[i] * AST + cc[i]];
      bf16* lb = &Bs[p][rr[i] * AST + cc[i]];
#if USE_ASYNC_LDS
      __builtin_amdgcn_global_load_async_to_lds_b128(
          (as1_v4i)(unsigned long long)ga,
          (as3_v4i)(unsigned int)(unsigned long long)la, 0, 0);
      __builtin_amdgcn_global_load_async_to_lds_b128(
          (as1_v4i)(unsigned long long)gb,
          (as3_v4i)(unsigned int)(unsigned long long)lb, 0, 0);
#else
      *(u32x4*)la = *(const u32x4*)ga;
      *(u32x4*)lb = *(const u32x4*)gb;
      __builtin_prefetch(ga + KT, 0, 0);   // global_prefetch_b8 of next K-tile
      __builtin_prefetch(gb + KT, 0, 0);
#endif
    }
  };

  v8f acc[4][2] = {};

  stage(0, 0);
  int p = 0;
  for (int kt = 0; kt < K; kt += KT, p ^= 1) {
#if USE_ASYNC_LDS
    wait_async0();                 // our stage-(p) async loads have landed
#endif
    __syncthreads();               // everyone's stage-(p) data visible / buffer p^1 released
    if (kt + KT < K) stage(p ^ 1, kt + KT);

    const bf16* Asp = &As[p][0];
    const bf16* Bsp = &Bs[p][0];
#pragma unroll
    for (int ks = 0; ks < 2; ++ks) {
      union U { v16bf v; bf16x8 h[2]; };
      U a[4], b[2];
      // A fragment: lane holds row (l16), K chunks {kb..kb+7} and {kb+16..kb+23}
#pragma unroll
      for (int tm = 0; tm < 4; ++tm) {
        int r  = wm * 64 + tm * 16 + l16;
        int kb = ks * 32 + hl * 8;
        a[tm].h[0] = *(const bf16x8*)(&Asp[r * AST + kb]);
        a[tm].h[1] = *(const bf16x8*)(&Asp[r * AST + kb + 16]);
      }
      // B fragment: lane holds weight-row (l16), K chunks {kb..kb+7} and {kb+16..kb+23}
#pragma unroll
      for (int tn = 0; tn < 2; ++tn) {
        int r  = wn * 32 + tn * 16 + l16;
        int kb = ks * 32 + hl * 8;
        b[tn].h[0] = *(const bf16x8*)(&Bsp[r * AST + kb]);
        b[tn].h[1] = *(const bf16x8*)(&Bsp[r * AST + kb + 16]);
      }
      // pin all 12 ds_load_b128 ahead of the 8 WMMAs so fragments get distinct
      // registers and waits become partial (load latency pipelined)
      sched_fence();
#pragma unroll
      for (int tm = 0; tm < 4; ++tm)
#pragma unroll
        for (int tn = 0; tn < 2; ++tn)
          acc[tm][tn] = __builtin_amdgcn_wmma_f32_16x16x32_bf16(
              false, a[tm].v, false, b[tn].v, (short)0, acc[tm][tn], false, false);
    }
  }

  // --- epilogue: VGPR v -> row v + 8*hl, col l16 ---
#pragma unroll
  for (int tm = 0; tm < 4; ++tm)
#pragma unroll
    for (int tn = 0; tn < 2; ++tn)
#pragma unroll
      for (int v = 0; v < 8; ++v) {
        int gr = bm + wm * 64 + tm * 16 + v + hl * 8;
        int gc = bn + wn * 32 + tn * 16 + l16;
        float val = acc[tm][tn][v];
        if (BIAS)  val += bias[gc];
        if (RESID) val += resid[(size_t)gr * N + gc];
        if (RELU)  val = val > 0.f ? val : 0.f;
        C[(size_t)gr * N + gc] = (OutT)val;
      }
}

// ---------------------------------------------------------------------------
// LN1: x = LayerNorm(concat(wbuf[row], tproj[tag]))  -> x_f32 and x_bf16
// one block (256 thr) per row of 1024; each thread owns 4 contiguous elems
// ---------------------------------------------------------------------------
__global__ __launch_bounds__(256)
void ln1_kernel(const float* __restrict__ wbuf, const float* __restrict__ tproj,
                const int* __restrict__ tag_ids,
                const float* __restrict__ g, const float* __restrict__ b,
                float* __restrict__ xf, bf16* __restrict__ xb) {
  int row = blockIdx.x, tid = threadIdx.x;
  int id = tag_ids[row];
  int trow = (id < 45) ? id : 45;          // row 45 == bt (uncovered)
  int j = tid * 4;
  float4 v = (j < 512)
      ? *(const float4*)(wbuf  + (size_t)row  * 512 + j)
      : *(const float4*)(tproj + (size_t)trow * 512 + (j - 512));
  float s = v.x + v.y + v.z + v.w;
  float q = v.x * v.x + v.y * v.y + v.z * v.z + v.w * v.w;
  for (int off = 16; off > 0; off >>= 1) {
    s += __shfl_xor(s, off, 32);
    q += __shfl_xor(q, off, 32);
  }
  __shared__ float ss[8], qq[8];
  int lane = tid & 31, wave = tid >> 5;
  if (lane == 0) { ss[wave] = s; qq[wave] = q; }
  __syncthreads();
  float st = 0.f, qt = 0.f;
  for (int w = 0; w < 8; ++w) { st += ss[w]; qt += qq[w]; }
  float mean = st * (1.f / 1024.f);
  float var  = qt * (1.f / 1024.f) - mean * mean;
  float r = rsqrtf(var + 1e-12f);
  float4 gv = *(const float4*)(g + j);
  float4 bv = *(const float4*)(b + j);
  float4 y;
  y.x = (v.x - mean) * r * gv.x + bv.x;
  y.y = (v.y - mean) * r * gv.y + bv.y;
  y.z = (v.z - mean) * r * gv.z + bv.z;
  y.w = (v.w - mean) * r * gv.w + bv.w;
  *(float4*)(xf + (size_t)row * 1024 + j) = y;
  bf16* o = xb + (size_t)row * 1024 + j;
  o[0] = (bf16)y.x; o[1] = (bf16)y.y; o[2] = (bf16)y.z; o[3] = (bf16)y.w;
}

// ---------------------------------------------------------------------------
// LN2: out = LayerNorm(z) (residual already fused into z by GEMM epilogue)
// ---------------------------------------------------------------------------
__global__ __launch_bounds__(256)
void ln2_kernel(const float* __restrict__ z,
                const float* __restrict__ g, const float* __restrict__ b,
                float* __restrict__ out) {
  int row = blockIdx.x, tid = threadIdx.x;
  int j = tid * 4;
  float4 v = *(const float4*)(z + (size_t)row * 1024 + j);
  float s = v.x + v.y + v.z + v.w;
  float q = v.x * v.x + v.y * v.y + v.z * v.z + v.w * v.w;
  for (int off = 16; off > 0; off >>= 1) {
    s += __shfl_xor(s, off, 32);
    q += __shfl_xor(q, off, 32);
  }
  __shared__ float ss[8], qq[8];
  int lane = tid & 31, wave = tid >> 5;
  if (lane == 0) { ss[wave] = s; qq[wave] = q; }
  __syncthreads();
  float st = 0.f, qt = 0.f;
  for (int w = 0; w < 8; ++w) { st += ss[w]; qt += qq[w]; }
  float mean = st * (1.f / 1024.f);
  float var  = qt * (1.f / 1024.f) - mean * mean;
  float r = rsqrtf(var + 1e-12f);
  float4 gv = *(const float4*)(g + j);
  float4 bv = *(const float4*)(b + j);
  float4 y;
  y.x = (v.x - mean) * r * gv.x + bv.x;
  y.y = (v.y - mean) * r * gv.y + bv.y;
  y.z = (v.z - mean) * r * gv.z + bv.z;
  y.w = (v.w - mean) * r * gv.w + bv.w;
  *(float4*)(out + (size_t)row * 1024 + j) = y;
}

// ---------------------------------------------------------------------------
// Host launcher
// ---------------------------------------------------------------------------
extern "C" void kernel_launch(void* const* d_in, const int* in_sizes, int n_in,
                              void* d_out, int out_size, void* d_ws, size_t ws_size,
                              hipStream_t stream) {
  (void)in_sizes; (void)n_in; (void)out_size; (void)ws_size;

  constexpr int Mtok = 8 * 2048;   // 16384 tokens
  constexpr int H = 1024, FF = 4096, HD = 512, T = 45;

  const float* word   = (const float*)d_in[0];
  const int*   tagids = (const int*)  d_in[1];
  const float* tagemb = (const float*)d_in[2];
  const float* Wv     = (const float*)d_in[3];
  const float* bv     = (const float*)d_in[4];
  const float* Wo     = (const float*)d_in[5];
  const float* bo     = (const float*)d_in[6];
  const float* Ww     = (const float*)d_in[7];
  const float* bw     = (const float*)d_in[8];
  const float* Wt     = (const float*)d_in[9];
  const float* bt     = (const float*)d_in[10];
  const float* W1     = (const float*)d_in[11];
  const float* b1     = (const float*)d_in[12];
  const float* W2     = (const float*)d_in[13];
  const float* b2     = (const float*)d_in[14];
  const float* ln1g   = (const float*)d_in[15];
  const float* ln1b   = (const float*)d_in[16];
  const float* ln2g   = (const float*)d_in[17];
  const float* ln2b   = (const float*)d_in[18];
  float* out = (float*)d_out;

  // workspace layout (zbuf aliases wordb+wbuf, both dead by the time ff2 runs)
  char* ws = (char*)d_ws;
  size_t off = 0;
  auto alloc = [&](size_t bytes) { void* p = ws + off; off += (bytes + 255) & ~size_t(255); return p; };
  bf16*  wordb = (bf16*) alloc((size_t)Mtok * H * 2);   // 33.5 MB  (dead after w-GEMM)
  float* wbuf  = (float*)alloc((size_t)Mtok * HD * 4);  // 33.5 MB  (dead after LN1)
  float* zbuf  = (float*)ws;                            // 67 MB, aliases the two above
  bf16*  Wwb   = (bf16*) alloc((size_t)HD * H * 2);
  bf16*  W1b   = (bf16*) alloc((size_t)FF * H * 2);
  bf16*  W2b   = (bf16*) alloc((size_t)H * FF * 2);
  float* vtmp  = (float*)alloc((size_t)T * H * 4);
  float* attn  = (float*)alloc((size_t)T * H * 4);
  float* tproj = (float*)alloc((size_t)46 * HD * 4);
  float* xf    = (float*)alloc((size_t)Mtok * H * 4);   // LN1 output, fp32 (residual)
  bf16*  xb    = (bf16*) alloc((size_t)Mtok * H * 2);   // LN1 output, bf16 (GEMM A)
  bf16*  hbuf  = (bf16*) alloc((size_t)Mtok * FF * 2);  // relu(FF1) in bf16

  auto blks4 = [](size_t n) { return (unsigned)((n / 4 + 255) / 256); };

  // 1) bf16 conversions
  cvt_f32_to_bf16<<<blks4((size_t)Mtok * H), 256, 0, stream>>>(word, wordb, Mtok * H / 4);
  cvt_f32_to_bf16<<<blks4((size_t)HD * H),   256, 0, stream>>>(Ww,   Wwb,   HD * H / 4);
  cvt_f32_to_bf16<<<blks4((size_t)FF * H),   256, 0, stream>>>(W1,   W1b,   FF * H / 4);
  cvt_f32_to_bf16<<<blks4((size_t)H * FF),   256, 0, stream>>>(W2,   W2b,   H * FF / 4);

  // 2) per-tag chain: v -> attn_vec -> tproj (row 45 = bt for uncovered tokens)
  small_mm<<<(T * H + 255) / 256, 256, 0, stream>>>(tagemb, Wv, bv, vtmp, T, T, H, H);
  small_mm<<<(T * H + 255) / 256, 256, 0, stream>>>(vtmp,   Wo, bo, attn, T, T, H, H);
  small_mm<<<(46 * HD + 255) / 256, 256, 0, stream>>>(attn, Wt, bt, tproj, T, 46, HD, H);

  // 3) w = word @ Ww^T + bw   [16384 x 512]
  gemm_bf16_wmma<float, true, false, false>
      <<<dim3(HD / BN, Mtok / BM), 256, 0, stream>>>(wordb, Wwb, bw, nullptr, wbuf, Mtok, HD, H);

  // 4) x = LN1(concat(w, tproj[tag]))  -> xf (fp32) + xb (bf16)
  ln1_kernel<<<Mtok, 256, 0, stream>>>(wbuf, tproj, tagids, ln1g, ln1b, xf, xb);

  // 5) h = relu(x @ W1^T + b1)   [16384 x 4096], bf16
  gemm_bf16_wmma<bf16, true, true, false>
      <<<dim3(FF / BN, Mtok / BM), 256, 0, stream>>>(xb, W1b, b1, nullptr, hbuf, Mtok, FF, H);

  // 6) z = h @ W2^T + b2 + x   [16384 x 1024], fp32 (residual fused)
  gemm_bf16_wmma<float, true, false, true>
      <<<dim3(H / BN, Mtok / BM), 256, 0, stream>>>(hbuf, W2b, b2, xf, zbuf, Mtok, H, FF);

  // 7) out = LN2(z)
  ln2_kernel<<<Mtok, 256, 0, stream>>>(zbuf, ln2g, ln2b, out);
}
